// target_attention_layer_67765993996881
// MI455X (gfx1250) — compile-verified, matrix operand order checked by reading, not run on hardware
//
#include <hip/hip_runtime.h>
#include <stdint.h>

// Problem constants (match reference)
#define BB   128
#define LQS  32
#define LKS  512
#define DM   1024
#define HID  1024
#define NH   16
#define HD   64
#define NEG_INF_F (-4294967295.0f)

typedef __attribute__((ext_vector_type(16))) _Float16 v16h;
typedef __attribute__((ext_vector_type(8)))  _Float16 v8h;
typedef __attribute__((ext_vector_type(4)))  _Float16 v4h;
typedef __attribute__((ext_vector_type(8)))  float    v8f;

// Build a 16-half A/B WMMA fragment: halves 0..7 = [base..base+7],
// halves 8..15 = [base+16..base+23] (per 16-bit A-matrix VGPR layout).
__device__ __forceinline__ v16h load_frag(const _Float16* base) {
  const v8h lo = *reinterpret_cast<const v8h*>(base);
  const v8h hi = *reinterpret_cast<const v8h*>(base + 16);
  v16h r;
#pragma unroll
  for (int i = 0; i < 8; ++i) { r[i] = lo[i]; r[i + 8] = hi[i]; }
  return r;
}

__device__ __forceinline__ v8f zero8() {
  v8f z;
#pragma unroll
  for (int i = 0; i < 8; ++i) z[i] = 0.0f;
  return z;
}

// ---------------- f32 -> f16 streaming convert (4 elems/thread) ----------------
__global__ __launch_bounds__(256) void cvt_f32_to_f16(const float* __restrict__ in,
                                                      _Float16* __restrict__ out) {
  const size_t i = ((size_t)blockIdx.x * blockDim.x + threadIdx.x) * 4;
  const float4 v = *reinterpret_cast<const float4*>(in + i);
  v4h o;
  o[0] = (_Float16)v.x; o[1] = (_Float16)v.y; o[2] = (_Float16)v.z; o[3] = (_Float16)v.w;
  *reinterpret_cast<v4h*>(out + i) = o;
}

// -------- weight convert + transpose: W[k][n] f32 -> Wt[n][k] f16 --------
__global__ __launch_bounds__(256) void wtrans(const float* __restrict__ W,
                                              _Float16* __restrict__ Wt) {
  const int n = blockIdx.x * 16 + threadIdx.x;
  const int k = blockIdx.y * 16 + threadIdx.y;
  Wt[(size_t)n * DM + k] = (_Float16)W[(size_t)k * HID + n];
}

// ---------------- fused GEMM (+bias, +head LayerNorm) ----------------
// A: [M, 1024] f16 row-major. Wt: [1024(n), 1024(k)] f16. Block = 4 waves on 4
// consecutive 16-row tiles of the SAME head, so the 64x32 B tile is shared:
// staged into LDS with double-buffered global_load_async_to_lds_b128.
// One barrier per k-step: a wave's ds reads of buffer `cur` complete (dscnt)
// before the WMMAs that consume them, which precede its next barrier signal,
// and refill of `cur` is only issued after that barrier completes collectively.
// Output f16: tr_out=0 -> [b,h,l,d]; tr_out=1 -> [b,h,d,l].
__global__ __launch_bounds__(128) void gemm_head_ln(
    const _Float16* __restrict__ A, const _Float16* __restrict__ Wt,
    const float* __restrict__ bias, const float* __restrict__ lnw,
    const float* __restrict__ lnb, _Float16* __restrict__ out,
    int L, int do_ln, int tr_out) {
  __shared__ __align__(16) _Float16 bstage[2][64 * 32];  // 2 x 4 KB

  const int tid  = threadIdx.x;
  const int wave = tid >> 5;
  const int lane = tid & 31;
  const int hi   = lane >> 4;
  const int l15  = lane & 15;
  const int m0   = (blockIdx.x * 4 + wave) * 16;
  const int head = blockIdx.y;
  const int n0   = head * HD;

  // Async-copy assignment: 256 16-byte chunks per stage, 2 per thread.
  // chunk c: B-row (n_local) = c>>2, k-col = (c&3)*8 halves.
  unsigned goff[2], lds_off[2];
#pragma unroll
  for (int j = 0; j < 2; ++j) {
    const int c   = tid * 2 + j;
    const int row = c >> 2;
    const int col = (c & 3) * 8;
    goff[j]    = ((unsigned)(n0 + row) * DM + (unsigned)col) * 2u;  // bytes
    lds_off[j] = ((unsigned)row * 32u + (unsigned)col) * 2u;        // bytes
  }
  const unsigned lds_base =
      (unsigned)(uintptr_t)(&bstage[0][0]);  // LDS aperture: low 32 bits = LDS byte addr
  const unsigned long long wt_base = (unsigned long long)(uintptr_t)Wt;

  auto issue_stage = [&](int buf, int k0) {
#pragma unroll
    for (int j = 0; j < 2; ++j) {
      const unsigned la = lds_base + (unsigned)buf * 4096u + lds_off[j];
      const unsigned ga = goff[j] + (unsigned)k0 * 2u;
      asm volatile("global_load_async_to_lds_b128 %0, %1, %2"
                   :: "v"(la), "v"(ga), "s"(wt_base)
                   : "memory");
    }
  };

  const _Float16* Arow = A + (size_t)(m0 + l15) * DM;

  v8f acc[4];
#pragma unroll
  for (int t = 0; t < 4; ++t) acc[t] = zero8();

  auto compute = [&](int buf, const v16h& a) {
    const _Float16* bb = &bstage[buf][0];
#pragma unroll
    for (int t = 0; t < 4; ++t) {
      const v16h b = load_frag(bb + (t * 16 + l15) * 32 + hi * 8);  // ds_load_b128
      acc[t] = __builtin_amdgcn_wmma_f32_16x16x32_f16(
          false, a, false, b, (short)0, acc[t], false, false);
    }
  };

  issue_stage(0, 0);
  v16h a0 = load_frag(Arow + hi * 8);
  v16h a1;

  // DM/32 = 32 k-steps, manually unrolled x2: even -> buf0/a0, odd -> buf1/a1.
  for (int kt = 0; kt < DM / 32; kt += 2) {
    // even phase: consume buf0 / a0, stage k-step kt+1 into buf1
    asm volatile("s_wait_asynccnt 0x0" ::: "memory");
    __syncthreads();
    issue_stage(1, (kt + 1) * 32);
    a1 = load_frag(Arow + (kt + 1) * 32 + hi * 8);
    __builtin_prefetch(Arow + (kt + 2) * 32, 0, 3);  // global_prefetch_b8
    compute(0, a0);

    // odd phase: consume buf1 / a1, stage k-step kt+2 into buf0
    asm volatile("s_wait_asynccnt 0x0" ::: "memory");
    __syncthreads();
    if (kt + 2 < DM / 32) {
      issue_stage(0, (kt + 2) * 32);
      a0 = load_frag(Arow + (kt + 2) * 32 + hi * 8);
      __builtin_prefetch(Arow + (kt + 3) * 32, 0, 3);
    }
    compute(1, a1);
  }

  float bt[4], lw[4], lb[4];
#pragma unroll
  for (int t = 0; t < 4; ++t) {
    bt[t] = bias[n0 + t * 16 + l15];
    lw[t] = lnw[t * 16 + l15];
    lb[t] = lnb[t * 16 + l15];
  }
#pragma unroll
  for (int t = 0; t < 4; ++t)
#pragma unroll
    for (int r = 0; r < 8; ++r) acc[t][r] += bt[t];

  if (do_ln) {
    // Row r+8*hi of the tile lives at VGPR r across a 16-lane half-wave;
    // xor masks 1..8 reduce within that half-wave only.
#pragma unroll
    for (int r = 0; r < 8; ++r) {
      float s = 0.f, s2 = 0.f;
#pragma unroll
      for (int t = 0; t < 4; ++t) { const float v = acc[t][r]; s += v; s2 += v * v; }
#pragma unroll
      for (int m = 1; m < 16; m <<= 1) {
        s  += __shfl_xor(s,  m, 32);
        s2 += __shfl_xor(s2, m, 32);
      }
      const float mu  = s * (1.0f / 64.0f);
      const float var = s2 * (1.0f / 64.0f) - mu * mu;
      const float inv = rsqrtf(var + 1e-12f);
#pragma unroll
      for (int t = 0; t < 4; ++t)
        acc[t][r] = (acc[t][r] - mu) * inv * lw[t] + lb[t];
    }
  }

#pragma unroll
  for (int t = 0; t < 4; ++t)
#pragma unroll
    for (int r = 0; r < 8; ++r) {
      const int gm = m0 + r + 8 * hi;
      const int bi = gm / L, l = gm % L;
      const int d  = t * 16 + l15;
      const size_t idx = tr_out
          ? ((size_t)(bi * NH + head) * HD + d) * (size_t)L + l
          : ((size_t)(bi * NH + head) * L + l) * HD + d;
      out[idx] = (_Float16)acc[t][r];
    }
}

// ---------------- fused attention: scores -> mask -> softmax -> att@V ----------------
__global__ __launch_bounds__(256) void attn_kernel(
    const _Float16* __restrict__ Qn, const _Float16* __restrict__ Kn,
    const _Float16* __restrict__ Vt, const int* __restrict__ seq,
    float* __restrict__ out, float* __restrict__ att) {
  __shared__ _Float16 att_lds[LQS * LKS];  // 32 KB
  __shared__ float red[2][LQS][4];

  const int bid  = blockIdx.x;     // h*B + b
  const int h    = bid >> 7;
  const int b    = bid & 127;
  const int wave = threadIdx.x >> 5;
  const int lane = threadIdx.x & 31;
  const int hi   = lane >> 4;
  const int l15  = lane & 15;
  const int rt   = wave & 1;       // q row-tile (16 rows)
  const int nc   = wave >> 1;      // 128-key chunk

  const _Float16* Qb = Qn + (size_t)((b * NH + h) * LQS) * HD;
  const _Float16* Kb = Kn + (size_t)((b * NH + h) * LKS) * HD;
  const _Float16* Vb = Vt + (size_t)((b * NH + h) * HD) * LKS;

  // ---- scores = Qn @ Kn^T ----
  v8f sc[8];
#pragma unroll
  for (int t = 0; t < 8; ++t) sc[t] = zero8();

#pragma unroll
  for (int ks = 0; ks < HD; ks += 32) {
    const int kb = ks + hi * 8;
    const v16h a = load_frag(Qb + (size_t)(rt * 16 + l15) * HD + kb);
#pragma unroll
    for (int t = 0; t < 8; ++t) {
      const int key = nc * 128 + t * 16 + l15;
      const v16h bv = load_frag(Kb + (size_t)key * HD + kb);
      sc[t] = __builtin_amdgcn_wmma_f32_16x16x32_f16(
          false, a, false, bv, (short)0, sc[t], false, false);
    }
  }

  // ---- scale + pad mask ----
#pragma unroll
  for (int t = 0; t < 8; ++t) {
    const int key = nc * 128 + t * 16 + l15;
    const bool keep = seq[b * LKS + key] != 0;
#pragma unroll
    for (int r = 0; r < 8; ++r)
      sc[t][r] = keep ? sc[t][r] * 0.125f : NEG_INF_F;
  }

  // ---- row max (partial per wave, then across key-chunks via LDS) ----
#pragma unroll
  for (int r = 0; r < 8; ++r) {
    float mx = sc[0][r];
#pragma unroll
    for (int t = 1; t < 8; ++t) mx = fmaxf(mx, sc[t][r]);
#pragma unroll
    for (int m = 1; m < 16; m <<= 1) mx = fmaxf(mx, __shfl_xor(mx, m, 32));
    if (l15 == 0) red[0][rt * 16 + r + 8 * hi][nc] = mx;
  }
  __syncthreads();

  // ---- exp + row sum ----
#pragma unroll
  for (int r = 0; r < 8; ++r) {
    const int q = rt * 16 + r + 8 * hi;
    const float mx = fmaxf(fmaxf(red[0][q][0], red[0][q][1]),
                           fmaxf(red[0][q][2], red[0][q][3]));
    float s = 0.f;
#pragma unroll
    for (int t = 0; t < 8; ++t) {
      const float e = __expf(sc[t][r] - mx);
      sc[t][r] = e;
      s += e;
    }
#pragma unroll
    for (int m = 1; m < 16; m <<= 1) s += __shfl_xor(s, m, 32);
    if (l15 == 0) red[1][q][nc] = s;
  }
  __syncthreads();

  // ---- normalize; write att (f32 -> d_out, f16 -> LDS) ----
  float* attb = att + (size_t)(h * BB + b) * LQS * LKS;
#pragma unroll
  for (int r = 0; r < 8; ++r) {
    const int q = rt * 16 + r + 8 * hi;
    const float inv = 1.0f / (red[1][q][0] + red[1][q][1] + red[1][q][2] + red[1][q][3]);
#pragma unroll
    for (int t = 0; t < 8; ++t) {
      const int key = nc * 128 + t * 16 + l15;
      const float p = sc[t][r] * inv;
      attb[(size_t)q * LKS + key] = p;
      att_lds[q * LKS + key] = (_Float16)p;
    }
  }
  __syncthreads();

  // ---- out = att @ V (waves repartition: 2 q-tiles x 4 d-tiles, full 512 keys) ----
  const int rt2 = wave & 1;
  const int dc  = wave >> 1;  // 0..3
  v8f oa = zero8();
#pragma unroll
  for (int kk = 0; kk < LKS; kk += 32) {
    const int kb = kk + hi * 8;
    const v16h a  = load_frag(att_lds + (rt2 * 16 + l15) * LKS + kb);  // ds_load_b128
    const v16h bv = load_frag(Vb + (size_t)(dc * 16 + l15) * LKS + kb);
    oa = __builtin_amdgcn_wmma_f32_16x16x32_f16(
        false, a, false, bv, (short)0, oa, false, false);
  }
  float* outb = out + (size_t)b * LQS * HID + h * HD;
#pragma unroll
  for (int r = 0; r < 8; ++r) {
    const int q = rt2 * 16 + r + 8 * hi;
    const int d = dc * 16 + l15;
    outb[(size_t)q * HID + d] = oa[r];
  }
}

extern "C" void kernel_launch(void* const* d_in, const int* in_sizes, int n_in,
                              void* d_out, int out_size, void* d_ws, size_t ws_size,
                              hipStream_t stream) {
  (void)in_sizes; (void)n_in; (void)out_size; (void)ws_size;
  const float* tq   = (const float*)d_in[0];
  const float* item = (const float*)d_in[1];
  const int*   seq  = (const int*)d_in[2];
  const float* Wq   = (const float*)d_in[3];
  const float* bq   = (const float*)d_in[4];
  const float* Wk   = (const float*)d_in[5];
  const float* bk   = (const float*)d_in[6];
  const float* Wv   = (const float*)d_in[7];
  const float* bv   = (const float*)d_in[8];
  const float* lnw  = (const float*)d_in[9];
  const float* lnb  = (const float*)d_in[10];

  float* out = (float*)d_out;
  float* att = out + (size_t)BB * LQS * HID;

  char* ws = (char*)d_ws;
  size_t off = 0;
  _Float16* tq_h = (_Float16*)(ws + off); off += (size_t)BB * LQS * DM * 2;
  _Float16* it_h = (_Float16*)(ws + off); off += (size_t)BB * LKS * DM * 2;
  _Float16* wq_t = (_Float16*)(ws + off); off += (size_t)DM * HID * 2;
  _Float16* wk_t = (_Float16*)(ws + off); off += (size_t)DM * HID * 2;
  _Float16* wv_t = (_Float16*)(ws + off); off += (size_t)DM * HID * 2;
  _Float16* qn   = (_Float16*)(ws + off); off += (size_t)BB * NH * LQS * HD * 2;
  _Float16* kn   = (_Float16*)(ws + off); off += (size_t)BB * NH * LKS * HD * 2;
  _Float16* vt   = (_Float16*)(ws + off); off += (size_t)BB * NH * HD * LKS * 2;

  const int nQ = BB * LQS * DM;   // 4,194,304
  const int nI = BB * LKS * DM;   // 67,108,864
  cvt_f32_to_f16<<<nQ / 4 / 256, 256, 0, stream>>>(tq, tq_h);
  cvt_f32_to_f16<<<nI / 4 / 256, 256, 0, stream>>>(item, it_h);

  dim3 tb(16, 16), tg(HID / 16, DM / 16);
  wtrans<<<tg, tb, 0, stream>>>(Wq, wq_t);
  wtrans<<<tg, tb, 0, stream>>>(Wk, wk_t);
  wtrans<<<tg, tb, 0, stream>>>(Wv, wv_t);

  gemm_head_ln<<<dim3((BB * LQS) / 64, NH), 128, 0, stream>>>(
      tq_h, wq_t, bq, lnw, lnb, qn, LQS, 1, 0);
  gemm_head_ln<<<dim3((BB * LKS) / 64, NH), 128, 0, stream>>>(
      it_h, wk_t, bk, lnw, lnb, kn, LKS, 1, 0);
  gemm_head_ln<<<dim3((BB * LKS) / 64, NH), 128, 0, stream>>>(
      it_h, wv_t, bv, lnw, lnb, vt, LKS, 0, 1);

  attn_kernel<<<NH * BB, 256, 0, stream>>>(qn, kn, vt, seq, out, att);
}
